// CTAttention_49349174231385
// MI455X (gfx1250) — compile-verified, matrix operand order checked by reading, not run on hardware
//
#include <hip/hip_runtime.h>
#include <hip/hip_bf16.h>

typedef __attribute__((ext_vector_type(16))) _Float16 v16h;
typedef __attribute__((ext_vector_type(8)))  _Float16 h8;
typedef __attribute__((ext_vector_type(4)))  _Float16 h4;
typedef __attribute__((ext_vector_type(8)))  float    v8f;
typedef __attribute__((ext_vector_type(4)))  float    f32x4;

#define BB 2
#define NN 1024
#define CC 512
#define HH 8
#define HD 64
#define KCLUS 8
#define MTOT (BB*NN)

union FragU { v16h v; h8 h[2]; };

__device__ __forceinline__ v8f wmma16(v16h a, v16h b, v8f c) {
    // D = A(16x32 f16) x B(32x16 f16) + C(16x16 f32)
    return __builtin_amdgcn_wmma_f32_16x16x32_f16(
        /*neg_a=*/false, a, /*neg_b=*/false, b,
        /*c_mod=*/(short)0, c, /*reuse_a=*/false, /*reuse_b=*/false);
}

// --------------------------------------------------------------------------
// Tensor Data Mover support (gfx1250).  6-arg builtin form (clang-23 /
// therock-10.0 headers): (u32x4 g0, i32x8 g1, i32x4 g2, i32x4 g3, i32x8, cpol)
// --------------------------------------------------------------------------
#if defined(__AMDGCN__) && __has_builtin(__builtin_amdgcn_tensor_load_to_lds)
#define HAVE_TDM 1
typedef __attribute__((ext_vector_type(4))) unsigned int u32x4;
typedef __attribute__((ext_vector_type(8))) int          i32x8;
typedef __attribute__((ext_vector_type(4))) int          i32x4;

// 2D tile load: tile (tile_d1 rows x tile_d0 elems) of 2-byte elements from
// a row-major tensor (row stride = stride0 elements) into LDS at lds_off.
__device__ __forceinline__ void tdm_load_2d(const _Float16* gaddr, unsigned lds_off,
                                            unsigned tensor_d0, unsigned tensor_d1,
                                            unsigned tile_d0, unsigned tile_d1,
                                            unsigned long stride0) {
    unsigned long ga = (unsigned long)(uintptr_t)gaddr;
    u32x4 g0;
    g0.x = 1u;                                              // count=1, user desc
    g0.y = lds_off;                                         // lds_addr (bytes)
    g0.z = (unsigned)ga;                                    // global_addr[31:0]
    g0.w = (unsigned)((ga >> 32) & 0x01FFFFFFu) | (2u << 30); // addr[56:32]|type=2
    i32x8 g1;
    g1[0] = (1 << 16);                                      // data_size = 2 bytes
    g1[1] = (int)((tensor_d0 & 0xFFFFu) << 16);             // tensor_dim0 lo
    g1[2] = (int)(((tensor_d0 >> 16) & 0xFFFFu) | ((tensor_d1 & 0xFFFFu) << 16));
    g1[3] = (int)(((tensor_d1 >> 16) & 0xFFFFu) | ((tile_d0 & 0xFFFFu) << 16));
    g1[4] = (int)(tile_d1 & 0xFFFFu);                       // tile_dim1, tile_dim2=0
    g1[5] = (int)(unsigned)(stride0 & 0xFFFFFFFFu);         // dim0_stride[31:0]
    g1[6] = (int)(unsigned)((stride0 >> 32) & 0xFFFFu);     // dim0_stride[47:32]
    g1[7] = 0;                                              // dim1_stride unused (2D)
    i32x4 z4 = {0, 0, 0, 0};
    i32x8 z8 = {0, 0, 0, 0, 0, 0, 0, 0};
    __builtin_amdgcn_tensor_load_to_lds(g0, g1, z4, z4, z8, 0);
}
#else
#define HAVE_TDM 0
#endif

// ---------------------------------------------------------------------------
// Zero-fill attn_map (512 MB): non-temporal b128 stores (region >> 192MB L2).
// ---------------------------------------------------------------------------
__global__ void ct_zero_kernel(f32x4* __restrict__ p, long n4) {
    long i = (long)blockIdx.x * blockDim.x + threadIdx.x;
    long stride = (long)gridDim.x * blockDim.x;
    f32x4 z = {0.f, 0.f, 0.f, 0.f};
    for (; i < n4; i += stride) __builtin_nontemporal_store(z, &p[i]);
}

// ---------------------------------------------------------------------------
// QKV projection: y = x @ W^T.  One wave computes one 16x16 tile via WMMA.
// grid = (MTOT/16, CC/16, 3)  -> z selects Q / K / V(transposed store)
// ---------------------------------------------------------------------------
__global__ __launch_bounds__(32)
void ct_qkv_kernel(const float* __restrict__ x,
                   const float* __restrict__ Wq,
                   const float* __restrict__ Wk,
                   const float* __restrict__ Wv,
                   _Float16* __restrict__ Qh,   // (B,H,N,hd)
                   _Float16* __restrict__ Kh,   // (B,H,N,hd)
                   _Float16* __restrict__ Vt) { // (B,H,hd,N)
    __shared__ _Float16 lA[16][32];
    __shared__ _Float16 lB[16][32];
    const int lane  = threadIdx.x;
    const int tm    = blockIdx.x;
    const int to    = blockIdx.y;
    const int which = blockIdx.z;
    const float* W = (which == 0) ? Wq : ((which == 1) ? Wk : Wv);

    v8f acc = {0.f,0.f,0.f,0.f,0.f,0.f,0.f,0.f};
    const int n = lane & 15, g = lane >> 4;

    for (int k0 = 0; k0 < CC; k0 += 32) {
        for (int t = 0; t < 4; ++t) {
            int p  = lane + 32 * t;
            int r  = p >> 3;
            int cq = (p & 7) * 4;
            f32x4 xv = *(const f32x4*)(x + (size_t)(tm * 16 + r) * CC + k0 + cq);
            h4 hx = { (_Float16)xv.x, (_Float16)xv.y, (_Float16)xv.z, (_Float16)xv.w };
            *(h4*)&lA[r][cq] = hx;
            f32x4 wv = *(const f32x4*)(W + (size_t)(to * 16 + r) * CC + k0 + cq);
            h4 hw = { (_Float16)wv.x, (_Float16)wv.y, (_Float16)wv.z, (_Float16)wv.w };
            *(h4*)&lB[r][cq] = hw;
        }
        __syncthreads();
        FragU a, bf;
        a.h[0]  = *(const h8*)&lA[n][8 * g];
        a.h[1]  = *(const h8*)&lA[n][16 + 8 * g];
        bf.h[0] = *(const h8*)&lB[n][8 * g];
        bf.h[1] = *(const h8*)&lB[n][16 + 8 * g];
        acc = wmma16(a.v, bf.v, acc);
        __syncthreads();
    }

    for (int r = 0; r < 8; ++r) {
        int m    = r + 8 * g;
        int row  = tm * 16 + m;
        int b    = row >> 10;
        int ntok = row & (NN - 1);
        int o    = to * 16 + n;
        int h    = o >> 6, d = o & 63;
        _Float16 val = (_Float16)acc[r];
        if (which == 0)
            Qh[(((size_t)(b * HH + h)) * NN + ntok) * HD + d] = val;
        else if (which == 1)
            Kh[(((size_t)(b * HH + h)) * NN + ntok) * HD + d] = val;
        else
            Vt[(((size_t)(b * HH + h)) * HD + d) * NN + ntok] = val;
    }
}

// ---------------------------------------------------------------------------
// Column sums of V: vsum[b,h,d] = sum_n V[b,h,n,d]  (Vt rows are contiguous)
// ---------------------------------------------------------------------------
__global__ void ct_vsum_kernel(const _Float16* __restrict__ Vt, float* __restrict__ vsum) {
    int t = blockIdx.x * blockDim.x + threadIdx.x;
    if (t < BB * HH * HD) {
        const _Float16* p = Vt + (size_t)t * NN;
        float s = 0.f;
        for (int i = 0; i < NN; ++i) s += (float)p[i];
        vsum[t] = s;
    }
}

// ---------------------------------------------------------------------------
// Fused masked attention.  K/V tiles staged in LDS by the Tensor Data Mover
// (double buffered, s_wait_tensorcnt), WMMA for S and e@V, NT scatter of the
// masked raw scores into attn_map.
// ---------------------------------------------------------------------------
__global__ __launch_bounds__(32)
void ct_attn_kernel(const _Float16* __restrict__ Qh,
                    const _Float16* __restrict__ Kh,
                    const _Float16* __restrict__ Vt,
                    const float*    __restrict__ vsum,
                    const int*      __restrict__ idx,
                    _Float16*       __restrict__ Oh,         // (B,N,C) head-major
                    float*          __restrict__ attn_map) { // (B,K,H,N,N)
    __shared__ _Float16 eT[16][32];
    __shared__ int ciS[16];
    __shared__ int cjS[32];
#if HAVE_TDM
    __shared__ _Float16 ldsK[2][32][64];  // rows j-local, cols d
    __shared__ _Float16 ldsV[2][64][32];  // rows d,       cols j-local
#endif

    const int lane = threadIdx.x;
    const int ti = blockIdx.x, h = blockIdx.y, b = blockIdx.z;
    const int i0 = ti * 16;
    const int n = lane & 15, g = lane >> 4;
    const float scale = 0.125f;               // hd^-0.5 = 1/8

    const _Float16* Qbh = Qh + ((size_t)(b * HH + h)) * NN * HD;
    const _Float16* Kbh = Kh + ((size_t)(b * HH + h)) * NN * HD;
    const _Float16* Vbh = Vt + ((size_t)(b * HH + h)) * HD * NN;

    if (lane < 16) ciS[lane] = idx[b * NN + i0 + lane];
    __syncthreads();

    // Q fragments for this i-tile, loaded once (contiguous b128 per chunk).
    FragU qa[2];
    for (int kc = 0; kc < 2; ++kc) {
        const _Float16* base = Qbh + (size_t)(i0 + n) * HD + 32 * kc;
        qa[kc].h[0] = *(const h8*)(base + 8 * g);
        qa[kc].h[1] = *(const h8*)(base + 16 + 8 * g);
    }

    v8f acc[4];
    float rs[8];
    for (int t = 0; t < 4; ++t)
        for (int r = 0; r < 8; ++r) acc[t][r] = 0.f;
    for (int r = 0; r < 8; ++r) rs[r] = 0.f;

#if HAVE_TDM
    // prologue: TDM-stage K/V tiles for j0=0 into buffer 0
    tdm_load_2d(Kbh, (unsigned)(uintptr_t)(void*)&ldsK[0][0][0],
                HD, NN, HD, 32, HD);
    tdm_load_2d(Vbh, (unsigned)(uintptr_t)(void*)&ldsV[0][0][0],
                NN, HD, 32, HD, NN);
#endif

    for (int j0 = 0; j0 < NN; j0 += 32) {
        cjS[lane] = idx[b * NN + j0 + lane];
#if HAVE_TDM
        const int buf = (j0 >> 5) & 1;
        if (j0 + 32 < NN) {
            // stage next chunk into the other buffer, then wait for current
            tdm_load_2d(Kbh + (size_t)(j0 + 32) * HD,
                        (unsigned)(uintptr_t)(void*)&ldsK[buf ^ 1][0][0],
                        HD, NN, HD, 32, HD);
            tdm_load_2d(Vbh + (j0 + 32),
                        (unsigned)(uintptr_t)(void*)&ldsV[buf ^ 1][0][0],
                        NN, HD, 32, HD, NN);
            __builtin_amdgcn_s_wait_tensorcnt(2);
        } else {
            __builtin_amdgcn_s_wait_tensorcnt(0);
        }
#endif
        __syncthreads();

        // two 16x16 score tiles covering j0..j0+31
        for (int half = 0; half < 2; ++half) {
            v8f st = {0.f,0.f,0.f,0.f,0.f,0.f,0.f,0.f};
            FragU kb;
            for (int kc = 0; kc < 2; ++kc) {
#if HAVE_TDM
                kb.h[0] = *(const h8*)&ldsK[buf][16 * half + n][32 * kc + 8 * g];
                kb.h[1] = *(const h8*)&ldsK[buf][16 * half + n][32 * kc + 16 + 8 * g];
#else
                const _Float16* base = Kbh + (size_t)(j0 + 16 * half + n) * HD + 32 * kc;
                kb.h[0] = *(const h8*)(base + 8 * g);
                kb.h[1] = *(const h8*)(base + 16 + 8 * g);
#endif
                st = wmma16(qa[kc].v, kb.v, st);
            }
            const int jl = 16 * half + n;
            const int cj = cjS[jl];
            for (int r = 0; r < 8; ++r) {
                int m = r + 8 * g;
                int i = i0 + m;
                int j = j0 + jl;
                float val = st[r] * scale;
                float e = 0.f;
                int ci = ciS[m];
                if (ci == cj) {
                    __builtin_nontemporal_store(
                        val, attn_map + ((((size_t)b * KCLUS + ci) * HH + h) * NN + i) * NN + j);
                    if (val != 0.f) e = __expf(val);  // exact-zero scores masked like ref
                }
                rs[r] += e;
                eT[m][jl] = (_Float16)e;
            }
        }
        __syncthreads();

        // acc += e(16x32) @ V(32x64): 4 WMMAs over hd chunks of 16
        FragU ea;
        ea.h[0] = *(const h8*)&eT[n][8 * g];
        ea.h[1] = *(const h8*)&eT[n][16 + 8 * g];
        for (int t = 0; t < 4; ++t) {
            FragU vb;
#if HAVE_TDM
            vb.h[0] = *(const h8*)&ldsV[buf][16 * t + n][8 * g];
            vb.h[1] = *(const h8*)&ldsV[buf][16 * t + n][16 + 8 * g];
#else
            const _Float16* base = Vbh + (size_t)(16 * t + n) * NN + j0;
            vb.h[0] = *(const h8*)(base + 8 * g);
            vb.h[1] = *(const h8*)(base + 16 + 8 * g);
#endif
            acc[t] = wmma16(ea.v, vb.v, acc[t]);
        }
        __syncthreads();
    }

    // reduce row sums across the 16 lanes that share each row
    for (int r = 0; r < 8; ++r) {
        float v = rs[r];
        v += __shfl_xor(v, 1, 32);
        v += __shfl_xor(v, 2, 32);
        v += __shfl_xor(v, 4, 32);
        v += __shfl_xor(v, 8, 32);
        rs[r] = v;
    }

    const float* vs = vsum + (size_t)(b * HH + h) * HD;
    const float epsN = 1e-6f / (float)NN;
    const float eps  = 1e-6f;
    for (int t = 0; t < 4; ++t) {
        for (int r = 0; r < 8; ++r) {
            int m = r + 8 * g;
            int d = 16 * t + n;
            float o = (acc[t][r] + epsN * vs[d]) / (rs[r] + eps);
            Oh[((size_t)(b * NN + i0 + m)) * CC + h * HD + d] = (_Float16)o;
        }
    }
}

// ---------------------------------------------------------------------------
// Output projection: out = Oh @ Wproj^T + bproj  (f32 result)
// ---------------------------------------------------------------------------
__global__ __launch_bounds__(32)
void ct_proj_kernel(const _Float16* __restrict__ Oh,
                    const float* __restrict__ Wp,
                    const float* __restrict__ bias,
                    float* __restrict__ out) {
    __shared__ _Float16 lB[16][32];
    const int lane = threadIdx.x;
    const int tm = blockIdx.x, to = blockIdx.y;
    const int n = lane & 15, g = lane >> 4;

    v8f acc = {0.f,0.f,0.f,0.f,0.f,0.f,0.f,0.f};
    for (int k0 = 0; k0 < CC; k0 += 32) {
        for (int t = 0; t < 4; ++t) {
            int p  = lane + 32 * t;
            int r  = p >> 3;
            int cq = (p & 7) * 4;
            f32x4 wv = *(const f32x4*)(Wp + (size_t)(to * 16 + r) * CC + k0 + cq);
            h4 hw = { (_Float16)wv.x, (_Float16)wv.y, (_Float16)wv.z, (_Float16)wv.w };
            *(h4*)&lB[r][cq] = hw;
        }
        __syncthreads();
        FragU a, bf;
        const _Float16* base = Oh + (size_t)(tm * 16 + n) * CC + k0;
        a.h[0]  = *(const h8*)(base + 8 * g);
        a.h[1]  = *(const h8*)(base + 16 + 8 * g);
        bf.h[0] = *(const h8*)&lB[n][8 * g];
        bf.h[1] = *(const h8*)&lB[n][16 + 8 * g];
        acc = wmma16(a.v, bf.v, acc);
        __syncthreads();
    }
    for (int r = 0; r < 8; ++r) {
        int m = r + 8 * g;
        int o = to * 16 + n;
        out[(size_t)(tm * 16 + m) * CC + o] = acc[r] + bias[o];
    }
}

// ---------------------------------------------------------------------------
extern "C" void kernel_launch(void* const* d_in, const int* in_sizes, int n_in,
                              void* d_out, int out_size, void* d_ws, size_t ws_size,
                              hipStream_t stream) {
    (void)in_sizes; (void)n_in; (void)out_size; (void)ws_size;
    const float* x_token = (const float*)d_in[0];
    // d_in[1] = x_path (unused by reference math)
    const int*   idx     = (const int*)d_in[2];
    // d_in[3] = cluster_num (fixed at 8)
    const float* Wq    = (const float*)d_in[4];
    const float* Wk    = (const float*)d_in[5];
    const float* Wv    = (const float*)d_in[6];
    const float* Wproj = (const float*)d_in[7];
    const float* bproj = (const float*)d_in[8];

    float* out      = (float*)d_out;
    float* attn_map = out + (size_t)BB * NN * CC;          // tuple output #2

    // workspace: Qh | Kh | Vt | Oh (each 2 MB f16) | vsum (4 KB f32)
    _Float16* Qh  = (_Float16*)d_ws;
    _Float16* Kh  = Qh + (size_t)BB * HH * NN * HD;
    _Float16* Vt  = Kh + (size_t)BB * HH * NN * HD;
    _Float16* Oh  = Vt + (size_t)BB * HH * NN * HD;
    float*    vsm = (float*)(Oh + (size_t)MTOT * CC);

    // 1) zero the 512 MB attn_map (dominant memory stage, NT stores)
    long n4 = (long)BB * KCLUS * HH * NN * NN / 4;
    ct_zero_kernel<<<16384, 256, 0, stream>>>((f32x4*)attn_map, n4);

    // 2) QKV projections (WMMA f16)
    ct_qkv_kernel<<<dim3(MTOT / 16, CC / 16, 3), 32, 0, stream>>>(
        x_token, Wq, Wk, Wv, Qh, Kh, Vt);

    // 3) V column sums for eps/N smoothing
    ct_vsum_kernel<<<dim3((BB * HH * HD + 255) / 256), 256, 0, stream>>>(Vt, vsm);

    // 4) fused masked attention + attn_map scatter (WMMA + TDM)
    ct_attn_kernel<<<dim3(NN / 16, HH, BB), 32, 0, stream>>>(
        Qh, Kh, Vt, vsm, idx, Oh, attn_map);

    // 5) output projection + bias (WMMA f16)
    ct_proj_kernel<<<dim3(MTOT / 16, CC / 16), 32, 0, stream>>>(Oh, Wproj, bproj, out);
}
